// MultiHeadAttention_38414187495921
// MI455X (gfx1250) — compile-verified, matrix operand order checked by reading, not run on hardware
//
#include <hip/hip_runtime.h>
#include <hip/hip_bf16.h>

// ---------------------------------------------------------------------------
// MultiHeadAttention with sliding-window band mask, CDNA5 (gfx1250) WMMA path.
// B=2, S=2048, D=1024, H=16, dk=64, band |i-j| <= 128.
// ---------------------------------------------------------------------------

typedef __attribute__((ext_vector_type(16))) __bf16 v16bf;
typedef __attribute__((ext_vector_type(8)))  float  v8f;
typedef float vf4 __attribute__((ext_vector_type(4)));
typedef int   vi4 __attribute__((vector_size(16)));   // matches async builtin param

#define D_MODEL 1024
#define S_LEN   2048
#define NHEAD   16
#define DK      64
#define BATCH   2
#define M_ROWS  (BATCH * S_LEN)         // 4096
#define BAND    128
#define WIN     272                     // 17 * 16 aligned band window
#define WPAD    288                     // 9 * 32 (K-chunk padded)
#define OUT_ELEMS (BATCH * S_LEN * D_MODEL)   // 4,194,304

#if defined(__gfx1250__) && \
    __has_builtin(__builtin_amdgcn_global_load_async_to_lds_b128) && \
    __has_builtin(__builtin_amdgcn_s_wait_asynccnt)
#define HAVE_ASYNC_LDS 1
#else
#define HAVE_ASYNC_LDS 0
#endif

// A-fragment element->local-K mapping for 16-bit 16x32 A tiles (ISA 7.12.2):
// lanes 0-15 : VGPR 0..3 -> K=0..7,  VGPR 4..7 -> K=16..23
// lanes 16-31: VGPR 0..3 -> K=8..15, VGPR 4..7 -> K=24..31
__device__ __forceinline__ int a_klocal(int lane, int pair) {
    int g = (lane >> 4) & 1;
    return (pair < 4) ? (g * 8 + 2 * pair) : (16 + g * 8 + 2 * (pair - 4));
}

// ---------------------------------------------------------------------------
// C[M x 1024] = X[M x 1024] @ W[1024 x 1024] + bias, bf16 WMMA, f32 accum.
// MODE 0: write row-major [M, N];  MODE 1: scatter to [B, H, S, dk].
// One wave per 16x64 output strip (A fragment reused across 4 N tiles).
// ---------------------------------------------------------------------------
template <int MODE>
__global__ __launch_bounds__(32)
void gemm_bias_wmma(const float* __restrict__ X,
                    const float* __restrict__ W,
                    const float* __restrict__ bias,
                    float* __restrict__ out) {
    const int lane = threadIdx.x;
    const int n0 = blockIdx.x * 64;
    const int m0 = blockIdx.y * 16;
    const float* xrow = X + (size_t)(m0 + (lane & 15)) * D_MODEL;

    v8f c[4] = {v8f{}, v8f{}, v8f{}, v8f{}};
    for (int k0 = 0; k0 < D_MODEL; k0 += 32) {
        v16bf a;
#pragma unroll
        for (int p = 0; p < 8; ++p) {
            const int kl = a_klocal(lane, p);
            const float* ap = xrow + k0 + kl;
            a[2 * p]     = (__bf16)ap[0];
            a[2 * p + 1] = (__bf16)ap[1];
        }
#pragma unroll
        for (int t = 0; t < 4; ++t) {
            const int ncol = n0 + t * 16 + (lane & 15);
            v16bf b;
#pragma unroll
            for (int p = 0; p < 8; ++p) {
                const int kb = k0 + a_klocal(lane, p);
                b[2 * p]     = (__bf16)W[(size_t)kb * D_MODEL + ncol];
                b[2 * p + 1] = (__bf16)W[(size_t)(kb + 1) * D_MODEL + ncol];
            }
            c[t] = __builtin_amdgcn_wmma_f32_16x16x32_bf16(false, a, false, b,
                                                           (short)0, c[t],
                                                           false, false);
        }
    }

    const int g = lane >> 4;
#pragma unroll
    for (int t = 0; t < 4; ++t) {
        const int n = n0 + t * 16 + (lane & 15);
        const float bi = bias[n];
#pragma unroll
        for (int v = 0; v < 8; ++v) {
            const int m = m0 + v + 8 * g;
            const float val = c[t][v] + bi;
            if (MODE == 0) {
                out[(size_t)m * D_MODEL + n] = val;
            } else {
                const int bb = m >> 11;          // / S_LEN
                const int ss = m & (S_LEN - 1);
                const int hh = n >> 6;           // / DK
                const int dd = n & (DK - 1);
                out[(((size_t)(bb * NHEAD + hh) * S_LEN) + ss) * DK + dd] = val;
            }
        }
    }
}

// ---------------------------------------------------------------------------
// Stage one 32-row x 64-col f32 chunk of the V window into LDS.
// Async path: CDNA5 global_load_async_to_lds_b128 (ASYNCcnt, 16 instrs/chunk).
// ---------------------------------------------------------------------------
__device__ __forceinline__ void stage_v_chunk(const float* __restrict__ vbase,
                                              int j0, int kc,
                                              float (*__restrict__ Vc)[DK],
                                              int lane) {
    const int g  = lane >> 4;          // row parity within a pair
    const int cg = (lane & 15) * 4;    // 16-byte column group
#pragma unroll
    for (int it = 0; it < 16; ++it) {
        const int r  = it * 2 + g;
        const int j  = j0 + kc * 32 + r;
        const int jc = min(max(j, 0), S_LEN - 1);      // clamp; P==0 there
        const float* gp = vbase + (size_t)jc * DK + cg;
#if HAVE_ASYNC_LDS
        __builtin_amdgcn_global_load_async_to_lds_b128(
            (vi4*)gp, (vi4*)&Vc[r][cg], 0, 0);
#else
        *(vf4*)&Vc[r][cg] = *(const vf4*)gp;
#endif
    }
}

// ---------------------------------------------------------------------------
// Band attention. One wave per (b, h, 16-row query tile).
// ---------------------------------------------------------------------------
__global__ __launch_bounds__(32)
void attn_band_wmma(const float* __restrict__ q,
                    const float* __restrict__ k,
                    const float* __restrict__ v,
                    float* __restrict__ p_attn,
                    float* __restrict__ xout) {
    __shared__ float P[16][WPAD];          // 18 KB: band scores -> probabilities
    __shared__ float Vs[2][32][DK];        // 16 KB: double-buffered V chunks

    const int lane = threadIdx.x;
    const int tile = blockIdx.x;
    const int qt = tile & 127;
    const int h  = (tile >> 7) & 15;
    const int b  = tile >> 11;
    const int q0 = qt * 16;
    const int j0 = q0 - BAND;

    const size_t hbase = (size_t)(b * NHEAD + h) * S_LEN * DK;
    const float* qbase = q + hbase;
    const float* kbase = k + hbase;
    const float* vbase = v + hbase;

    const int m = lane & 15;
    const int g = lane >> 4;

    // ---- Q tile A-fragments (16x64 = two 16x32 halves) ----
    v16bf aq[2];
#pragma unroll
    for (int kh = 0; kh < 2; ++kh) {
#pragma unroll
        for (int p = 0; p < 8; ++p) {
            const int kl = kh * 32 + a_klocal(lane, p);
            const float* ap = qbase + (size_t)(q0 + m) * DK + kl;
            aq[kh][2 * p]     = (__bf16)ap[0];
            aq[kh][2 * p + 1] = (__bf16)ap[1];
        }
    }

    // ---- scores: S = Q K^T / sqrt(dk) over 17 key tiles ----
    for (int kt = 0; kt < 17; ++kt) {
        const int krow = j0 + kt * 16 + (lane & 15);
        const int kc = min(max(krow, 0), S_LEN - 1);
        v8f c = {};
#pragma unroll
        for (int kh = 0; kh < 2; ++kh) {
            v16bf bb;
#pragma unroll
            for (int p = 0; p < 8; ++p) {
                const int kl = kh * 32 + a_klocal(lane, p);
                const float* bp = kbase + (size_t)kc * DK + kl;
                bb[2 * p]     = (__bf16)bp[0];
                bb[2 * p + 1] = (__bf16)bp[1];
            }
            c = __builtin_amdgcn_wmma_f32_16x16x32_bf16(false, aq[kh], false, bb,
                                                        (short)0, c, false, false);
        }
        const int n = lane & 15;
#pragma unroll
        for (int vv = 0; vv < 8; ++vv) {
            const int mm = vv + 8 * g;
            const int i  = q0 + mm;
            const int j  = j0 + kt * 16 + n;
            const bool ok = (j >= 0) && (j < S_LEN) &&
                            (j >= i - BAND) && (j <= i + BAND);
            P[mm][kt * 16 + n] = ok ? (c[vv] * 0.125f) : -1.0e9f;
        }
    }
    __syncthreads();

    // ---- softmax over the 272 window cols (2 lanes per row) ----
    {
        const int mm = lane & 15;
        const int c0 = (lane >> 4) * 136;
        float mx = -3.0e38f;
        for (int ci = 0; ci < 136; ++ci) mx = fmaxf(mx, P[mm][c0 + ci]);
        mx = fmaxf(mx, __shfl_xor(mx, 16, 32));
        float sum = 0.0f;
        for (int ci = 0; ci < 136; ++ci) {
            const float e = __expf(P[mm][c0 + ci] - mx);   // masked -> 0 (underflow)
            P[mm][c0 + ci] = e;
            sum += e;
        }
        sum += __shfl_xor(sum, 16, 32);
        const float inv = 1.0f / sum;
        for (int ci = 0; ci < 136; ++ci) P[mm][c0 + ci] *= inv;
        if ((lane >> 4) == 1)                       // zero K-chunk padding tail
            for (int ci = WIN; ci < WPAD; ++ci) P[mm][ci] = 0.0f;
    }
    __syncthreads();

    // ---- stream full p_attn rows: vf4 non-temporal stores (HBM-bound) ----
    const size_t prow_base = ((size_t)(b * NHEAD + h) * S_LEN + q0) * S_LEN;
    for (int mm = 0; mm < 16; ++mm) {
        float* prow = p_attn + prow_base + (size_t)mm * S_LEN;
        for (int c4 = lane * 4; c4 < S_LEN; c4 += 128) {
            vf4 val;
            const int w0 = c4 - j0;
            val.x = (w0 + 0 >= 0 && w0 + 0 < WIN) ? P[mm][w0 + 0] : 0.0f;
            val.y = (w0 + 1 >= 0 && w0 + 1 < WIN) ? P[mm][w0 + 1] : 0.0f;
            val.z = (w0 + 2 >= 0 && w0 + 2 < WIN) ? P[mm][w0 + 2] : 0.0f;
            val.w = (w0 + 3 >= 0 && w0 + 3 < WIN) ? P[mm][w0 + 3] : 0.0f;
            __builtin_nontemporal_store(val, (vf4*)(prow + c4));
        }
    }

    // ---- context: X = P @ V, V chunks staged (async) into LDS, dbl-buffered ----
    v8f cx[4] = {v8f{}, v8f{}, v8f{}, v8f{}};
    stage_v_chunk(vbase, j0, 0, Vs[0], lane);
    for (int kc = 0; kc < 9; ++kc) {
        if (kc < 8) stage_v_chunk(vbase, j0, kc + 1, Vs[(kc + 1) & 1], lane);
#if HAVE_ASYNC_LDS
        if (kc < 8) __builtin_amdgcn_s_wait_asynccnt(16);   // chunk kc landed
        else        __builtin_amdgcn_s_wait_asynccnt(0);
#endif
        const float (*Vc)[DK] = Vs[kc & 1];
        v16bf ap;
#pragma unroll
        for (int p = 0; p < 8; ++p) {
            const int kl = a_klocal(lane, p);
            ap[2 * p]     = (__bf16)P[m][kc * 32 + kl];
            ap[2 * p + 1] = (__bf16)P[m][kc * 32 + kl + 1];
        }
#pragma unroll
        for (int nt = 0; nt < 4; ++nt) {
            const int nc = nt * 16 + (lane & 15);
            v16bf bp;
#pragma unroll
            for (int p = 0; p < 8; ++p) {
                const int kl = a_klocal(lane, p);
                bp[2 * p]     = (__bf16)Vc[kl][nc];
                bp[2 * p + 1] = (__bf16)Vc[kl + 1][nc];
            }
            cx[nt] = __builtin_amdgcn_wmma_f32_16x16x32_bf16(false, ap, false, bp,
                                                             (short)0, cx[nt],
                                                             false, false);
        }
    }
#pragma unroll
    for (int nt = 0; nt < 4; ++nt) {
        const int n = nt * 16 + (lane & 15);
#pragma unroll
        for (int vv = 0; vv < 8; ++vv) {
            const int i = q0 + vv + 8 * g;
            xout[(size_t)(b * S_LEN + i) * D_MODEL + h * DK + n] = cx[nt][vv];
        }
    }
}

// ---------------------------------------------------------------------------
extern "C" void kernel_launch(void* const* d_in, const int* in_sizes, int n_in,
                              void* d_out, int out_size, void* d_ws, size_t ws_size,
                              hipStream_t stream) {
    (void)in_sizes; (void)n_in; (void)out_size; (void)ws_size;

    const float* query = (const float*)d_in[0];
    const float* key_  = (const float*)d_in[1];
    const float* value = (const float*)d_in[2];
    const float* Wq = (const float*)d_in[3];  const float* bq = (const float*)d_in[4];
    const float* Wk = (const float*)d_in[5];  const float* bk = (const float*)d_in[6];
    const float* Wv = (const float*)d_in[7];  const float* bv = (const float*)d_in[8];
    const float* Wo = (const float*)d_in[9];  const float* bo = (const float*)d_in[10];

    float* out    = (float*)d_out;            // [B,S,D]
    float* p_attn = out + OUT_ELEMS;          // [B,H,S,S]

    float* ws = (float*)d_ws;                 // 4 x 16 MB scratch
    float* qb = ws;
    float* kb = ws + (size_t)OUT_ELEMS;
    float* vb = ws + (size_t)2 * OUT_ELEMS;
    float* xb = ws + (size_t)3 * OUT_ELEMS;

    const dim3 gemm_grid(D_MODEL / 64, M_ROWS / 16);   // 16 x 256 strips
    const dim3 wave(32);

    gemm_bias_wmma<1><<<gemm_grid, wave, 0, stream>>>(query, Wq, bq, qb);
    gemm_bias_wmma<1><<<gemm_grid, wave, 0, stream>>>(key_,  Wk, bk, kb);
    gemm_bias_wmma<1><<<gemm_grid, wave, 0, stream>>>(value, Wv, bv, vb);

    attn_band_wmma<<<BATCH * NHEAD * (S_LEN / 16), wave, 0, stream>>>(qb, kb, vb,
                                                                      p_attn, xb);

    gemm_bias_wmma<0><<<gemm_grid, wave, 0, stream>>>(xb, Wo, bo, out);
}
